// VecLocal2d_47390669144708
// MI455X (gfx1250) — matrix operand order repaired and checked by reading, not compiled, not run on hardware
//
#include <hip/hip_runtime.h>

// ---------------------------------------------------------------------------
// VecLocal2d on MI455X (gfx1250): per-pixel block-diagonal GEMM via
// v_wmma_f32_16x16x32_bf16 with bf16 hi/lo split for ~fp32 accuracy.
//   M=160 (B*K), N=64 (C_OUT), Kdim=288 (C_IN*3*3), 1024 pixels.
// ---------------------------------------------------------------------------

typedef __attribute__((ext_vector_type(16))) __bf16 v16bf;
typedef __attribute__((ext_vector_type(8)))  float  v8f;

#define HO     32
#define WO     32
#define CIN    32
#define COUT   64
#define MROWS  160            // B*K = 16*10
#define FDIM   288            // CIN * 9
#define PITCH  40             // LDS row pitch in ushorts (80B: conflict-free b128)

#define SB_ELEMS (9 * COUT * PITCH)      // 23040 ushorts per plane
#define SA_ELEMS (MROWS * PITCH)         //  6400 ushorts per plane
#define SMEM_BYTES ((2 * SB_ELEMS + 2 * SA_ELEMS) * 2)   // 117,760 B

struct alignas(16) U4 { unsigned int x0, x1, x2, x3; };
union Frag { U4 q[2]; v16bf v; };

__device__ __forceinline__ unsigned short bf16_hi(float f) {
  unsigned int u = __float_as_uint(f);
  u += 0x7FFFu + ((u >> 16) & 1u);      // round-to-nearest-even to bf16
  return (unsigned short)(u >> 16);
}
__device__ __forceinline__ float bf16_f(unsigned short h) {
  return __uint_as_float(((unsigned int)h) << 16);
}

extern "C" __global__ __launch_bounds__(256)
void veclocal2d_wmma(const float* __restrict__ x,     // [160][32][32][32]
                     const float* __restrict__ wgt,   // [32][32][64][32][3][3]
                     const float* __restrict__ bias,  // [10][64][32][32]
                     float* __restrict__ out) {       // [160][64][32][32]
  extern __shared__ unsigned short smem[];
  unsigned short* sBhi = smem;                 // [tap][o][PITCH]
  unsigned short* sBlo = sBhi + SB_ELEMS;
  unsigned short* sAhi = sBlo + SB_ELEMS;      // [m][PITCH]
  unsigned short* sAlo = sAhi + SA_ELEMS;

  const int tid  = threadIdx.x;
  const int lane = tid & 31;                   // wave32
  const int wave = tid >> 5;                   // 8 waves
  const int pix  = blockIdx.x;                 // output pixel
  const int i    = pix >> 5;
  const int j    = pix & 31;

  // ---- Stage this pixel's weight slab [COUT][FDIM] once (coalesced b128),
  //      split into bf16 hi/lo, layout [tap][o][c]. 18 float4 per thread. ----
  {
    const U4* wp = reinterpret_cast<const U4*>(wgt + (size_t)pix * (COUT * FDIM));
    #pragma unroll
    for (int q = 0; q < 18; ++q) {
      int idx4 = q * 256 + tid;
      U4 raw = wp[idx4];
      float v4[4] = { __uint_as_float(raw.x0), __uint_as_float(raw.x1),
                      __uint_as_float(raw.x2), __uint_as_float(raw.x3) };
      int e = idx4 * 4;
      #pragma unroll
      for (int u = 0; u < 4; ++u) {
        int ee  = e + u;
        int o   = ee / FDIM;
        int r   = ee - o * FDIM;               // = c*9 + tap
        int c   = r / 9;
        int tap = r - c * 9;
        float f = v4[u];
        unsigned short hi = bf16_hi(f);
        unsigned short lo = bf16_hi(f - bf16_f(hi));
        int idx = (tap * COUT + o) * PITCH + c;
        sBhi[idx] = hi;
        sBlo[idx] = lo;
      }
    }
  }
  __syncthreads();

  v8f acc[5] = {};                              // 5 16x16 tiles per wave
  const int khalf = (lane >> 4) & 1;

  for (int tap = 0; tap < 9; ++tap) {
    int kh = tap / 3, kw = tap - kh * 3;
    int yi = i + kh - 1;
    int xj = j + kw - 1;
    bool valid = (yi >= 0) && (yi < 32) && (xj >= 0) && (xj < 32);
    int yoff = yi * 32 + xj;

    // ---- Stage A chunk [160][32] for this tap (L2-resident gathers) ----
    #pragma unroll
    for (int e = 0; e < 20; ++e) {
      int f = e * 256 + tid;                    // flat (m,c), f = m*32 + c
      float v = valid ? x[(size_t)f * 1024 + yoff] : 0.0f;
      unsigned short hi = bf16_hi(v);
      unsigned short lo = bf16_hi(v - bf16_f(hi));
      int m = f >> 5, c = f & 31;
      sAhi[m * PITCH + c] = hi;
      sAlo[m * PITCH + c] = lo;
    }
    __syncthreads();

    // ---- 5 tiles per wave: 3 split-precision WMMAs each ----
    #pragma unroll
    for (int t = 0; t < 5; ++t) {
      int id = wave * 5 + t;                    // 0..39
      int mT = id >> 2, nT = id & 3;

      // A fragment 16x32 bf16: lane-half -> K {0..15} vs {16..31}
      int mrow = mT * 16 + (lane & 15);
      const U4* pah = reinterpret_cast<const U4*>(sAhi + mrow * PITCH + khalf * 16);
      const U4* pal = reinterpret_cast<const U4*>(sAlo + mrow * PITCH + khalf * 16);
      Frag ahi, alo;
      ahi.q[0] = pah[0]; ahi.q[1] = pah[1];     // +16B = K+8
      alo.q[0] = pal[0]; alo.q[1] = pal[1];

      // B fragment 32x16 bf16: lane-half -> K {0..15} vs {16..31}
      int col = nT * 16 + (lane & 15);
      const U4* pbh = reinterpret_cast<const U4*>(sBhi + (tap * COUT + col) * PITCH + khalf * 16);
      const U4* pbl = reinterpret_cast<const U4*>(sBlo + (tap * COUT + col) * PITCH + khalf * 16);
      Frag bhi, blo;
      bhi.q[0] = pbh[0]; bhi.q[1] = pbh[1];
      blo.q[0] = pbl[0]; blo.q[1] = pbl[1];

      // fp32 ~= hi*hi + hi*lo + lo*hi, accumulated in f32
      acc[t] = __builtin_amdgcn_wmma_f32_16x16x32_bf16(false, ahi.v, false, bhi.v,
                                                       (short)0, acc[t], false, false);
      acc[t] = __builtin_amdgcn_wmma_f32_16x16x32_bf16(false, ahi.v, false, blo.v,
                                                       (short)0, acc[t], false, false);
      acc[t] = __builtin_amdgcn_wmma_f32_16x16x32_bf16(false, alo.v, false, bhi.v,
                                                       (short)0, acc[t], false, false);
    }
    __syncthreads();                            // before next tap rewrites sA
  }

  // ---- Epilogue: C layout (VGPR r -> M = r + 8*laneHalf), add bias, store ----
  #pragma unroll
  for (int t = 0; t < 5; ++t) {
    int id = wave * 5 + t;
    int mT = id >> 2, nT = id & 3;
    int col   = nT * 16 + (lane & 15);
    int mbase = mT * 16 + ((lane >> 4) << 3);
    #pragma unroll
    for (int r = 0; r < 8; ++r) {
      int bk   = mbase + r;                     // 0..159 (= b*10 + k)
      int kcat = bk % 10;
      float v = acc[t][r] + bias[((size_t)(kcat * COUT + col)) * 1024 + pix];
      out[((size_t)(bk * COUT + col)) * 1024 + pix] = v;
    }
  }
}

extern "C" void kernel_launch(void* const* d_in, const int* in_sizes, int n_in,
                              void* d_out, int out_size, void* d_ws, size_t ws_size,
                              hipStream_t stream) {
  (void)in_sizes; (void)n_in; (void)d_ws; (void)ws_size; (void)out_size;
  const float* x   = (const float*)d_in[0];
  const float* w   = (const float*)d_in[1];
  const float* b   = (const float*)d_in[2];
  float*       out = (float*)d_out;

  // 117,760 B dynamic LDS per block (> default 64KB cap): opt in.
  hipFuncSetAttribute((const void*)veclocal2d_wmma,
                      hipFuncAttributeMaxDynamicSharedMemorySize, SMEM_BYTES);

  veclocal2d_wmma<<<HO * WO, 256, SMEM_BYTES, stream>>>(x, w, b, out);
}
